// CodeCoveragePredictionModel_71451075937029
// MI455X (gfx1250) — matrix operand order repaired and checked by reading, not compiled, not run on hardware
//
#include <hip/hip_runtime.h>
#include <math.h>

#define Bsz  64
#define Tlen 512
#define Hdim 768
#define Snum 128

typedef __attribute__((ext_vector_type(16))) __bf16 v16bf;
typedef __attribute__((ext_vector_type(8)))  float  v8f;

union FragU { v16bf v; uint4 q[2]; };

__device__ __forceinline__ unsigned short f32_to_bf16_u(float x) {
  unsigned int u = __builtin_bit_cast(unsigned int, x);
  unsigned int lsb = (u >> 16) & 1u;
  u += 0x7fffu + lsb;                 // round-to-nearest-even
  return (unsigned short)(u >> 16);
}
__device__ __forceinline__ float bf16_u_to_f32(unsigned short h) {
  unsigned int u = ((unsigned int)h) << 16;
  return __builtin_bit_cast(float, u);
}
__device__ __forceinline__ float gelu_exact(float x) {
  return 0.5f * x * (1.0f + erff(x * 0.70710678118654752f));
}

// ---------------------------------------------------------------------------
// Kernel 1: pack w1/w2 (f32, row-major [K][N]) into WMMA B-fragment order,
// split into bf16 hi + lo.  Fragment layout (per 32x16 K-tile x N-tile):
//   lane 0-15 : N = nt*16+lane, elements j<8 -> K = kt*32 + j, j>=8 -> K = kt*32 + 16 + (j-8)
//   lane16-31 : N = nt*16+lane-16, j<8 -> K = kt*32 + 8 + j, j>=8 -> K = kt*32 + 24 + (j-8)
// packed index e = ((kt*48 + nt)*32 + lane)*16 + j
// ---------------------------------------------------------------------------
__global__ void pack_weights_kernel(const float* __restrict__ w1,
                                    const float* __restrict__ w2,
                                    unsigned short* __restrict__ w1h,
                                    unsigned short* __restrict__ w1l,
                                    unsigned short* __restrict__ w2h,
                                    unsigned short* __restrict__ w2l) {
  int e = blockIdx.x * blockDim.x + threadIdx.x;     // 0 .. 2*H*H-1
  const int HH = Hdim * Hdim;
  int which = e / HH;
  int idx   = e - which * HH;
  int j    = idx & 15;
  int t    = idx >> 4;
  int lane = t & 31;
  int t2   = t >> 5;
  int nt   = t2 % 48;
  int kt   = t2 / 48;
  int K = kt * 32 + ((j < 8) ? (((lane & 16) ? 8 : 0) + j) : (16 + ((lane & 16) ? 8 : 0) + (j - 8)));
  int N = nt * 16 + (lane & 15);
  const float* w = which ? w2 : w1;
  float v = w[K * Hdim + N];
  unsigned short hu = f32_to_bf16_u(v);
  float rem = v - bf16_u_to_f32(hu);
  unsigned short lu = f32_to_bf16_u(rem);
  if (which) { w2h[idx] = hu; w2l[idx] = lu; }
  else       { w1h[idx] = hu; w1l[idx] = lu; }
}

// ---------------------------------------------------------------------------
// Kernel 2: per-(batch,statement) mean pooling (ids are sorted -> binary
// search for the token range), output split into bf16 hi/lo rows [B*S, H].
// ---------------------------------------------------------------------------
__global__ void pool_split_kernel(const float* __restrict__ hidden,
                                  const int*   __restrict__ sid,
                                  unsigned short* __restrict__ Xh,
                                  unsigned short* __restrict__ Xl) {
  int r = blockIdx.x;          // r = b*S + s
  int b = r >> 7;
  int s = r & 127;
  const int* row = sid + b * Tlen;
  int lo = 0, hi = Tlen;
  while (lo < hi) { int mid = (lo + hi) >> 1; if (row[mid] <  s) lo = mid + 1; else hi = mid; }
  int start = lo;
  lo = start; hi = Tlen;
  while (lo < hi) { int mid = (lo + hi) >> 1; if (row[mid] <= s) lo = mid + 1; else hi = mid; }
  int end = lo;
  float inv = (end > start) ? 1.0f / (float)(end - start) : 0.0f;
  const float* hb = hidden + (size_t)b * Tlen * Hdim;
  for (int h = threadIdx.x; h < Hdim; h += blockDim.x) {
    float sum = 0.0f;
    for (int t = start; t < end; ++t) sum += hb[(size_t)t * Hdim + h];
    float m = sum * inv;
    unsigned short hu = f32_to_bf16_u(m);
    float rem = m - bf16_u_to_f32(hu);
    unsigned short lu = f32_to_bf16_u(rem);
    size_t o = (size_t)r * Hdim + h;
    Xh[o] = hu; Xl[o] = lu;
  }
}

// ---------------------------------------------------------------------------
// Kernel 3: fused 3-layer MLP. 8 waves / block, 32 rows / block.
// bf16x3 split GEMMs (AhBh + AhBl + AlBh) on v_wmma_f32_16x16x32_bf16.
// LDS: XH|XL (96KB) + Y1H|Y1L (96KB); Y2(f32) aliases XH|XL.
// ---------------------------------------------------------------------------
__global__ __launch_bounds__(256, 1)
void mlp_wmma_kernel(const unsigned short* __restrict__ Xh,
                     const unsigned short* __restrict__ Xl,
                     const unsigned short* __restrict__ w1h,
                     const unsigned short* __restrict__ w1l,
                     const unsigned short* __restrict__ w2h,
                     const unsigned short* __restrict__ w2l,
                     const float* __restrict__ b1,
                     const float* __restrict__ b2,
                     const float* __restrict__ w3,
                     const float* __restrict__ b3,
                     float* __restrict__ out) {
  extern __shared__ char smem[];
  unsigned short* sXh  = (unsigned short*)smem;          // 32*768 u16
  unsigned short* sXl  = sXh  + 32 * Hdim;
  unsigned short* sY1h = sXl  + 32 * Hdim;
  unsigned short* sY1l = sY1h + 32 * Hdim;
  float*          sY2  = (float*)smem;                   // aliases sXh+sXl

  const int rbase = blockIdx.x * 32;
  const int tid  = threadIdx.x;
  const int wave = tid >> 5;
  const int lane = tid & 31;

  // ---- Phase 0: stage X tile (hi/lo) into LDS -----------------------------
  {
    const uint4* gh = (const uint4*)(Xh + (size_t)rbase * Hdim);
    const uint4* gl = (const uint4*)(Xl + (size_t)rbase * Hdim);
    uint4* shh = (uint4*)sXh;
    uint4* sll = (uint4*)sXl;
    const int total = 32 * Hdim / 8;                     // 3072 uint4
    for (int i = tid; i < total; i += 256) { shh[i] = gh[i]; sll[i] = gl[i]; }
  }
  __syncthreads();

  const int m_sub = wave & 1;                            // 2 row sub-tiles
  const int n_grp = wave >> 1;                           // 4 column groups
  const int rowA  = m_sub * 16 + (lane & 15);            // A-frag row
  const int csel  = (lane & 16) ? 8 : 0;                 // A-frag K phase
  const int ncolL = lane & 15;
  const int mrow0 = m_sub * 16 + ((lane & 16) ? 8 : 0);  // C-frag row base

  // ---- Phase 1: Y1 = gelu(X @ w1 + b1) ------------------------------------
  for (int i = 0; i < 12; ++i) {
    int nt = n_grp + 4 * i;
    v8f acc = {0.f, 0.f, 0.f, 0.f, 0.f, 0.f, 0.f, 0.f};
    for (int kt = 0; kt < 24; ++kt) {
      int c0 = kt * 32 + csel;
      FragU ah, al, bh, bl;
      ah.q[0] = *(const uint4*)(sXh + rowA * Hdim + c0);
      ah.q[1] = *(const uint4*)(sXh + rowA * Hdim + c0 + 16);
      al.q[0] = *(const uint4*)(sXl + rowA * Hdim + c0);
      al.q[1] = *(const uint4*)(sXl + rowA * Hdim + c0 + 16);
      const uint4* ph = (const uint4*)w1h + ((size_t)(kt * 48 + nt) * 32 + lane) * 2;
      const uint4* pl = (const uint4*)w1l + ((size_t)(kt * 48 + nt) * 32 + lane) * 2;
      bh.q[0] = ph[0]; bh.q[1] = ph[1];
      bl.q[0] = pl[0]; bl.q[1] = pl[1];
      acc = __builtin_amdgcn_wmma_f32_16x16x32_bf16(false, ah.v, false, bh.v, (short)0, acc, false, false);
      acc = __builtin_amdgcn_wmma_f32_16x16x32_bf16(false, ah.v, false, bl.v, (short)0, acc, false, false);
      acc = __builtin_amdgcn_wmma_f32_16x16x32_bf16(false, al.v, false, bh.v, (short)0, acc, false, false);
    }
    int  ncol = nt * 16 + ncolL;
    float bias = b1[ncol];
#pragma unroll
    for (int r = 0; r < 8; ++r) {
      float y = gelu_exact(acc[r] + bias);
      unsigned short hu = f32_to_bf16_u(y);
      float rem = y - bf16_u_to_f32(hu);
      unsigned short lu = f32_to_bf16_u(rem);
      int m = mrow0 + r;
      sY1h[m * Hdim + ncol] = hu;
      sY1l[m * Hdim + ncol] = lu;
    }
  }
  __syncthreads();

  // ---- Phase 2: Y2 = gelu(Y1 @ w2 + b2) (f32 into aliased LDS) ------------
  for (int i = 0; i < 12; ++i) {
    int nt = n_grp + 4 * i;
    v8f acc = {0.f, 0.f, 0.f, 0.f, 0.f, 0.f, 0.f, 0.f};
    for (int kt = 0; kt < 24; ++kt) {
      int c0 = kt * 32 + csel;
      FragU ah, al, bh, bl;
      ah.q[0] = *(const uint4*)(sY1h + rowA * Hdim + c0);
      ah.q[1] = *(const uint4*)(sY1h + rowA * Hdim + c0 + 16);
      al.q[0] = *(const uint4*)(sY1l + rowA * Hdim + c0);
      al.q[1] = *(const uint4*)(sY1l + rowA * Hdim + c0 + 16);
      const uint4* ph = (const uint4*)w2h + ((size_t)(kt * 48 + nt) * 32 + lane) * 2;
      const uint4* pl = (const uint4*)w2l + ((size_t)(kt * 48 + nt) * 32 + lane) * 2;
      bh.q[0] = ph[0]; bh.q[1] = ph[1];
      bl.q[0] = pl[0]; bl.q[1] = pl[1];
      acc = __builtin_amdgcn_wmma_f32_16x16x32_bf16(false, ah.v, false, bh.v, (short)0, acc, false, false);
      acc = __builtin_amdgcn_wmma_f32_16x16x32_bf16(false, ah.v, false, bl.v, (short)0, acc, false, false);
      acc = __builtin_amdgcn_wmma_f32_16x16x32_bf16(false, al.v, false, bh.v, (short)0, acc, false, false);
    }
    int  ncol = nt * 16 + ncolL;
    float bias = b2[ncol];
#pragma unroll
    for (int r = 0; r < 8; ++r) {
      float y = gelu_exact(acc[r] + bias);
      sY2[(mrow0 + r) * Hdim + ncol] = y;
    }
  }
  __syncthreads();

  // ---- Phase 3: out = sigmoid(Y2 @ w3 + b3), wave reduction ----------------
  {
    float bb = b3[0];
    for (int m = wave * 4; m < wave * 4 + 4; ++m) {
      float p = 0.0f;
      for (int h = lane; h < Hdim; h += 32) p += sY2[m * Hdim + h] * w3[h];
      for (int off = 16; off > 0; off >>= 1) p += __shfl_xor(p, off, 32);
      if (lane == 0) out[rbase + m] = 1.0f / (1.0f + expf(-(p + bb)));
    }
  }
}

// ---------------------------------------------------------------------------
extern "C" void kernel_launch(void* const* d_in, const int* in_sizes, int n_in,
                              void* d_out, int out_size, void* d_ws, size_t ws_size,
                              hipStream_t stream) {
  const float* hidden = (const float*)d_in[0];
  const int*   sid    = (const int*)  d_in[1];
  const float* w1     = (const float*)d_in[2];
  const float* b1     = (const float*)d_in[3];
  const float* w2     = (const float*)d_in[4];
  const float* b2     = (const float*)d_in[5];
  const float* w3     = (const float*)d_in[6];
  const float* b3     = (const float*)d_in[7];
  float* out = (float*)d_out;

  char* ws = (char*)d_ws;
  const size_t szX = (size_t)Bsz * Snum * Hdim * sizeof(unsigned short); // 12.58 MB
  const size_t szW = (size_t)Hdim * Hdim * sizeof(unsigned short);       // 1.18 MB
  unsigned short* Xh  = (unsigned short*)(ws);
  unsigned short* Xl  = (unsigned short*)(ws + szX);
  unsigned short* w1h = (unsigned short*)(ws + 2 * szX);
  unsigned short* w1l = (unsigned short*)(ws + 2 * szX + szW);
  unsigned short* w2h = (unsigned short*)(ws + 2 * szX + 2 * szW);
  unsigned short* w2l = (unsigned short*)(ws + 2 * szX + 3 * szW);

  pack_weights_kernel<<<(2 * Hdim * Hdim) / 256, 256, 0, stream>>>(
      w1, w2, w1h, w1l, w2h, w2l);
  pool_split_kernel<<<Bsz * Snum, 128, 0, stream>>>(hidden, sid, Xh, Xl);
  mlp_wmma_kernel<<<(Bsz * Snum) / 32, 256, 192 * 1024, stream>>>(
      Xh, Xl, w1h, w1l, w2h, w2l, b1, b2, w3, b3, out);
}
